// Hierarchical_Generator_90675349553819
// MI455X (gfx1250) — matrix operand order repaired and checked by reading, not compiled, not run on hardware
//
#include <hip/hip_runtime.h>
#include <hip/hip_bf16.h>

// ---------------------------------------------------------------------------
// Problem constants (from reference)
// ---------------------------------------------------------------------------
#define NNODES 50000
#define NEDGE  800000
#define XLD    161      // x row stride: NFEAT(128) + EXTRA(32) + 1
#define LD     128      // common row stride for all N x * scratch matrices
#define KSEL   71       // NN + 1

typedef __attribute__((ext_vector_type(16))) __bf16 v16bf;
typedef __attribute__((ext_vector_type(8)))  __bf16 v8bf;
typedef __attribute__((ext_vector_type(8)))  float  v8f;

// ---------------------------------------------------------------------------
// Monotonic float <-> uint32 key (for atomicMin / radix select)
// ---------------------------------------------------------------------------
__device__ __forceinline__ unsigned encf(float f) {
  unsigned u = __float_as_uint(f);
  return (u & 0x80000000u) ? ~u : (u | 0x80000000u);
}
__device__ __forceinline__ float decf(unsigned k) {
  return __uint_as_float((k & 0x80000000u) ? (k & 0x7FFFFFFFu) : ~k);
}

// ---------------------------------------------------------------------------
// Utility kernels
// ---------------------------------------------------------------------------
__global__ void fill_f32(float* __restrict__ p, float v, long n) {
  long i = (long)blockIdx.x * blockDim.x + threadIdx.x;
  if (i < n) p[i] = v;
}
__global__ void set_u32(unsigned* p, unsigned v) { *p = v; }

// ---------------------------------------------------------------------------
// Pack a K x Nout f32 weight matrix into per-lane bf16 WMMA B fragments.
// Fragment layout for V_WMMA_F32_16X16X32_BF16 B (32x16):
//   half e (0..15) of lane L holds W[kt*32 + (e>>3)*16 + (L>>4)*8 + (e&7), nt*16 + (L&15)]
// Linear packed index: ((kt*ntiles + nt)*32 + lane)*16 + e  -> each lane's
// fragment is 16 contiguous halves (32 B): a single coalesced load.
// ---------------------------------------------------------------------------
__global__ void pack_w(const float* __restrict__ W, int K, int Nout,
                       __bf16* __restrict__ Wp)
{
  long i = (long)blockIdx.x * blockDim.x + threadIdx.x;
  long total = (long)K * Nout;
  if (i >= total) return;
  int ntiles = Nout >> 4;
  int e    = (int)(i & 15);
  int lane = (int)((i >> 4) & 31);
  long t2  = i >> 9;
  int nt   = (int)(t2 % ntiles);
  int kt   = (int)(t2 / ntiles);
  int Kg = kt * 32 + (e >> 3) * 16 + (lane >> 4) * 8 + (e & 7);
  int Ng = nt * 16 + (lane & 15);
  Wp[i] = (__bf16)W[(long)Kg * Nout + Ng];
}

// x[:, :128] -> bf16 copy with stride LD
__global__ void cvt_x_bf16(const float* __restrict__ x, __bf16* __restrict__ xb,
                           int Nr)
{
  long i = (long)blockIdx.x * blockDim.x + threadIdx.x;
  if (i >= (long)Nr * 128) return;
  int r = (int)(i >> 7), f = (int)(i & 127);
  xb[i] = (__bf16)x[(long)r * XLD + f];
}

// ---------------------------------------------------------------------------
// bf16 WMMA GEMM:  Out[M x (NT*16)] = A[M x K] (bf16) * Wpacked
// One wave per 16-row strip; per K-step: load A fragment once, run NT
// back-to-back v_wmma into NT accumulators.
// A-frag: lane<16 -> M=lane&15, halves e: K = kt*32 + (e&8)*2? --
//   concretely: chunk0 = A[row, kt*32 + hi*8 + 0..7],
//               chunk1 = A[row, kt*32 + hi*8 + 16..23]  (hi = lane>>4)
// C/D f32: VGPR r -> M = r + hi*8, N = nt*16 + (lane&15)
// ---------------------------------------------------------------------------
template <int NT>
__global__ __launch_bounds__(128)
void gemm_bf16_wmma(const __bf16* __restrict__ A, int lda,
                    const __bf16* __restrict__ Wp,
                    const float* __restrict__ bias,
                    float* __restrict__ OutF, __bf16* __restrict__ OutB,
                    int ldo, int M, int K, int fuse_relu)
{
  int wave = blockIdx.x * (blockDim.x >> 5) + (threadIdx.x >> 5);
  int nStrips = (M + 15) >> 4;
  if (wave >= nStrips) return;           // wave-uniform: EXEC full at wmma

  int lane = threadIdx.x & 31;
  int l15  = lane & 15;
  int hi   = lane >> 4;
  int arow = wave * 16 + l15;
  if (arow >= M) arow = M - 1;           // M % 16 == 0 in practice
  const __bf16* aRow = A + (long)arow * lda + hi * 8;
  const __bf16* wLane = Wp + (long)lane * 16;

  v8f acc[NT];
#pragma unroll
  for (int nt = 0; nt < NT; ++nt) acc[nt] = (v8f){};

  int ksteps = K >> 5;
  for (int kt = 0; kt < ksteps; ++kt) {
    const __bf16* ap = aRow + kt * 32;
    __builtin_prefetch(ap + 32, 0, 1);   // global_prefetch_b8 next K chunk
    v8bf a0 = *(const v8bf*)(ap);
    v8bf a1 = *(const v8bf*)(ap + 16);
    v16bf a = __builtin_shufflevector(a0, a1, 0, 1, 2, 3, 4, 5, 6, 7,
                                      8, 9, 10, 11, 12, 13, 14, 15);
#pragma unroll
    for (int nt = 0; nt < NT; ++nt) {
      v16bf b = *(const v16bf*)(wLane + ((long)kt * NT + nt) * 512);
      acc[nt] = __builtin_amdgcn_wmma_f32_16x16x32_bf16(
          false, a, false, b, (short)0, acc[nt], false, false);
    }
  }

  int mbase = wave * 16 + hi * 8;
#pragma unroll
  for (int nt = 0; nt < NT; ++nt) {
    int n = nt * 16 + l15;
    float bb = bias ? bias[n] : 0.0f;
#pragma unroll
    for (int r = 0; r < 8; ++r) {
      int mrow = mbase + r;
      if (mrow < M) {
        float t = acc[nt][r] + bb;
        if (fuse_relu) t = fmaxf(t, 0.0f);
        if (OutF) OutF[(long)mrow * ldo + n] = t;
        if (OutB) OutB[(long)mrow * ldo + n] = (__bf16)t;
      }
    }
  }
}

// ---------------------------------------------------------------------------
// Edge aggregation: agg[row[e], :] += val[e] * s[col[e], :]   (one wave/edge)
// s (19 MB) lives in L2 (192 MB) -> gathers & atomics run at L2 bandwidth.
// ---------------------------------------------------------------------------
__global__ __launch_bounds__(256)
void edge_aggregate(const float* __restrict__ s, int ld,
                    const int* __restrict__ row, const int* __restrict__ col,
                    const float* __restrict__ val,
                    float* __restrict__ agg, int E, int F)
{
  int e = blockIdx.x * (blockDim.x >> 5) + (threadIdx.x >> 5);
  if (e >= E) return;
  int lane = threadIdx.x & 31;
  int r = row[e], c = col[e];
  float v = val[e];
  const float* sp = s + (long)c * ld;
  float* ap = agg + (long)r * ld;
  for (int f = lane; f < F; f += 32)
    atomicAdd(ap + f, v * sp[f]);
}

// ---------------------------------------------------------------------------
// h = relu(agg + b); accumulate per-column sum / sumsq (for BN).
// ---------------------------------------------------------------------------
__global__ __launch_bounds__(128)
void bias_relu_stats(const float* __restrict__ agg, int ld,
                     const float* __restrict__ bias,
                     float* __restrict__ h, int ldh,
                     float* __restrict__ colsum, float* __restrict__ colsumsq,
                     int Nr, int F, int rowsPerBlock, int doStats)
{
  int f = threadIdx.x;
  if (f >= F) return;
  int r0 = blockIdx.x * rowsPerBlock;
  int r1 = min(r0 + rowsPerBlock, Nr);
  float bb = bias[f];
  float s = 0.0f, sq = 0.0f;
  for (int r = r0; r < r1; ++r) {
    float t = fmaxf(agg[(long)r * ld + f] + bb, 0.0f);
    h[(long)r * ldh + f] = t;
    s += t; sq += t * t;
  }
  if (doStats) {
    atomicAdd(&colsum[f], s);
    atomicAdd(&colsumsq[f], sq);
  }
}

// BN normalize; writes the bf16 activation buffer consumed by the next GEMM.
__global__ void bn_normalize(const float* __restrict__ h, int ld,
                             const float* __restrict__ colsum,
                             const float* __restrict__ colsumsq,
                             __bf16* __restrict__ hb,
                             int Nr, int F)
{
  long i = (long)blockIdx.x * blockDim.x + threadIdx.x;
  if (i >= (long)Nr * F) return;
  int r = (int)(i / F), f = (int)(i - (long)r * F);
  float m = colsum[f] / (float)Nr;
  float v = colsumsq[f] / (float)Nr - m * m;
  float t = h[(long)r * ld + f];
  hb[(long)r * ld + f] = (__bf16)((t - m) * rsqrtf(v + 1e-5f));
}

// feat = concat(h5[:, :64], x[:, 128:160]) -> bf16 (GEMM input)
__global__ void build_feat(const float* __restrict__ h5, int ldh,
                           const float* __restrict__ x, int ldx,
                           __bf16* __restrict__ featb, int ldf, int Nr)
{
  long i = (long)blockIdx.x * blockDim.x + threadIdx.x;
  if (i >= (long)Nr * 96) return;
  int r = (int)(i / 96), f = (int)(i - (long)r * 96);
  float t = (f < 64) ? h5[(long)r * ldh + f] : x[(long)r * ldx + 128 + (f - 64)];
  featb[(long)r * ldf + f] = (__bf16)t;
}

// m[r] = dot(m2[r, :64], w) + b  (one wave per row, shuffle reduction)
__global__ __launch_bounds__(256)
void head_dot64(const float* __restrict__ m2, int ld,
                const float* __restrict__ w, const float* __restrict__ b,
                float* __restrict__ m, int Nr)
{
  int r = blockIdx.x * (blockDim.x >> 5) + (threadIdx.x >> 5);
  if (r >= Nr) return;
  int lane = threadIdx.x & 31;
  const float* p = m2 + (long)r * ld;
  float t = p[lane] * w[lane] + p[lane + 32] * w[lane + 32];
#pragma unroll
  for (int off = 16; off > 0; off >>= 1) t += __shfl_xor(t, off, 32);
  if (lane == 0) m[r] = t + b[0];
}

__global__ void min_reduce(const float* __restrict__ m, int Nr, unsigned* key) {
  int i = blockIdx.x * blockDim.x + threadIdx.x;
  if (i >= Nr) return;
  atomicMin(key, encf(m[i]));
}

__global__ void replace_grp0(float* __restrict__ m,
                             const float* __restrict__ x, int ldx,
                             const unsigned* __restrict__ minkey, int Nr) {
  int i = blockIdx.x * blockDim.x + threadIdx.x;
  if (i >= Nr) return;
  if (x[(long)i * ldx + 160] == 0.0f) m[i] = decf(*minkey);
}

// ---------------------------------------------------------------------------
// k-th largest via 4-pass MSB radix select, single workgroup (1024 threads).
// ---------------------------------------------------------------------------
__global__ __launch_bounds__(1024)
void topk_select(const float* __restrict__ m, int Nr, int k,
                 float* __restrict__ thresh_out)
{
  __shared__ unsigned hist[256];
  __shared__ unsigned prefix_s;
  __shared__ int k_s;
  unsigned prefix = 0;
  int kk = k;
  for (int pass = 3; pass >= 0; --pass) {
    for (int b = threadIdx.x; b < 256; b += blockDim.x) hist[b] = 0;
    __syncthreads();
    unsigned mask = (pass == 3) ? 0u : (0xFFFFFFFFu << ((pass + 1) * 8));
    for (int i = threadIdx.x; i < Nr; i += blockDim.x) {
      unsigned key = encf(m[i]);
      if ((key & mask) == (prefix & mask))
        atomicAdd(&hist[(key >> (pass * 8)) & 0xFFu], 1u);
    }
    __syncthreads();
    if (threadIdx.x == 0) {
      int rem = kk, chosen = 0;
      for (int b = 255; b >= 0; --b) {
        int c = (int)hist[b];
        if (rem - c <= 0) { chosen = b; break; }
        rem -= c;
      }
      prefix_s = prefix | ((unsigned)chosen << (pass * 8));
      k_s = rem;
    }
    __syncthreads();
    prefix = prefix_s;
    kk = k_s;
    __syncthreads();
  }
  if (threadIdx.x == 0) *thresh_out = decf(prefix);
}

__global__ void finalize(const float* __restrict__ m,
                         const float* __restrict__ thresh,
                         float* __restrict__ out, int Nr)
{
  int i = blockIdx.x * blockDim.x + threadIdx.x;
  if (i >= Nr) return;
  float t = *thresh;
  float v = m[i];
  out[i] = (v > t) ? v * (1.0f / v) : 0.0f;
}

// ---------------------------------------------------------------------------
// Orchestration
// ---------------------------------------------------------------------------
extern "C" void kernel_launch(void* const* d_in, const int* in_sizes, int n_in,
                              void* d_out, int out_size, void* d_ws, size_t ws_size,
                              hipStream_t stream)
{
  const float* x   = (const float*)d_in[0];
  const int*   row = (const int*)  d_in[1];
  const int*   col = (const int*)  d_in[2];
  const float* val = (const float*)d_in[3];
  const float* W1 = (const float*)d_in[4];  const float* b1 = (const float*)d_in[5];
  const float* W2 = (const float*)d_in[6];  const float* b2 = (const float*)d_in[7];
  const float* W3 = (const float*)d_in[8];  const float* b3 = (const float*)d_in[9];
  const float* W4 = (const float*)d_in[10]; const float* b4 = (const float*)d_in[11];
  const float* W5 = (const float*)d_in[12]; const float* b5 = (const float*)d_in[13];
  const float* M1w = (const float*)d_in[14]; const float* M1b = (const float*)d_in[15];
  const float* M2w = (const float*)d_in[16]; const float* M2b = (const float*)d_in[17];
  const float* M3w = (const float*)d_in[18]; const float* M3b = (const float*)d_in[19];

  const int N = NNODES, E = NEDGE;

  // ---- workspace carve-up (all chunk sizes are multiples of 32 B) ----
  char* base = (char*)d_ws;
  auto take = [&](long bytes) { char* p = base; base += bytes; return p; };
  float*  h     = (float*) take((long)N * LD * 4);   // pre-BN / layer5 act (f32)
  float*  sbuf  = (float*) take((long)N * LD * 4);   // GEMM f32 out / m2
  float*  agg   = (float*) take((long)N * LD * 4);   // edge aggregation
  __bf16* xb    = (__bf16*)take((long)N * LD * 2);   // bf16 x features
  __bf16* hb    = (__bf16*)take((long)N * LD * 2);   // bf16 activations
  __bf16* featb = (__bf16*)take((long)N * LD * 2);   // bf16 MLP feat
  __bf16* m1b   = (__bf16*)take((long)N * LD * 2);   // bf16 m1
  __bf16* Wp[7];
  for (int i = 0; i < 7; ++i) Wp[i] = (__bf16*)take(16384 * 2);
  float*  mbuf  = (float*) take((long)N * 4);
  float*  colsum   = (float*) take(128 * 4);
  float*  colsumsq = (float*) take(128 * 4);
  unsigned* minkey = (unsigned*)take(32);
  float*  thresh   = (float*) take(32);

  // ---- pack all weights to bf16 B-fragment layout; convert x ----
  auto packw = [&](const float* W, int K, int Nout, __bf16* dst) {
    long tot = (long)K * Nout;
    pack_w<<<(int)((tot + 255) / 256), 256, 0, stream>>>(W, K, Nout, dst);
  };
  packw(W1, 128, 96, Wp[0]);
  packw(W2, 96, 96, Wp[1]);
  packw(W3, 96, 96, Wp[2]);
  packw(W4, 96, 96, Wp[3]);
  packw(W5, 96, 64, Wp[4]);
  packw(M1w, 96, 128, Wp[5]);
  packw(M2w, 128, 64, Wp[6]);
  cvt_x_bf16<<<(int)(((long)N * 128 + 255) / 256), 256, 0, stream>>>(x, xb, N);

  auto launch_gemm = [&](const __bf16* A, const __bf16* Wpk, const float* bias,
                         float* OutF, __bf16* OutB, int K, int Nout, int relu) {
    int strips = (N + 15) / 16;
    int blocks = (strips + 3) / 4;   // 4 waves / block
    switch (Nout >> 4) {
      case 8: gemm_bf16_wmma<8><<<blocks, 128, 0, stream>>>(A, LD, Wpk, bias,
                  OutF, OutB, LD, N, K, relu); break;
      case 6: gemm_bf16_wmma<6><<<blocks, 128, 0, stream>>>(A, LD, Wpk, bias,
                  OutF, OutB, LD, N, K, relu); break;
      default: gemm_bf16_wmma<4><<<blocks, 128, 0, stream>>>(A, LD, Wpk, bias,
                  OutF, OutB, LD, N, K, relu); break;
    }
  };

  auto gc_layer = [&](const __bf16* Ain, const __bf16* Wpk, const float* b,
                      int K, int F, int doBN) {
    launch_gemm(Ain, Wpk, nullptr, sbuf, nullptr, K, F, 0);
    long naz = (long)N * LD;
    fill_f32<<<(int)((naz + 255) / 256), 256, 0, stream>>>(agg, 0.0f, naz);
    fill_f32<<<1, 256, 0, stream>>>(colsum, 0.0f, 256);  // colsum+colsumsq
    edge_aggregate<<<(E + 7) / 8, 256, 0, stream>>>(sbuf, LD, row, col, val,
                                                    agg, E, F);
    const int rpb = 128;
    bias_relu_stats<<<(N + rpb - 1) / rpb, 128, 0, stream>>>(
        agg, LD, b, h, LD, colsum, colsumsq, N, F, rpb, doBN);
    if (doBN) {
      long tot = (long)N * F;
      bn_normalize<<<(int)((tot + 255) / 256), 256, 0, stream>>>(
          h, LD, colsum, colsumsq, hb, N, F);
    }
  };

  // 5 graph-conv layers (bf16 activations feed each GEMM)
  gc_layer(xb, Wp[0], b1, 128, 96, 1);
  gc_layer(hb, Wp[1], b2,  96, 96, 1);
  gc_layer(hb, Wp[2], b3,  96, 96, 1);
  gc_layer(hb, Wp[3], b4,  96, 96, 1);
  gc_layer(hb, Wp[4], b5,  96, 64, 0);   // layer5: relu only, h stays f32

  // MLP head
  long tot96 = (long)N * 96;
  build_feat<<<(int)((tot96 + 255) / 256), 256, 0, stream>>>(h, LD, x, XLD,
                                                             featb, LD, N);
  launch_gemm(featb, Wp[5], M1b, nullptr, m1b, 96, 128, 1);  // m1 (bf16 only)
  launch_gemm(m1b,   Wp[6], M2b, sbuf, nullptr, 128, 64, 1); // m2 (f32)
  head_dot64<<<(N + 7) / 8, 256, 0, stream>>>(sbuf, LD, M3w, M3b, mbuf, N);

  // min / group replacement / top-k threshold / final mask
  set_u32<<<1, 1, 0, stream>>>(minkey, 0xFFFFFFFFu);
  min_reduce<<<(N + 255) / 256, 256, 0, stream>>>(mbuf, N, minkey);
  replace_grp0<<<(N + 255) / 256, 256, 0, stream>>>(mbuf, x, XLD, minkey, N);
  topk_select<<<1, 1024, 0, stream>>>(mbuf, N, KSEL, thresh);
  finalize<<<(N + 255) / 256, 256, 0, stream>>>(mbuf, thresh, (float*)d_out, N);
}
